// TSPDEncoder_31542239822223
// MI455X (gfx1250) — matrix operand order repaired and checked by reading, not compiled, not run on hardware
//
#include <hip/hip_runtime.h>

// ---------------- problem constants ----------------
#define Dm     128          // D_MODEL
#define NH     8            // heads
#define DK     16           // d_k
#define FF     512          // D_FF
#define NL     3            // layers
#define NB     32           // buckets
#define Nn     1024         // nodes
#define N1     1025         // nodes + global
#define MP     1056         // padded rows (66 tiles of 16)
#define MT     66           // MP/16
#define Bsz    2
#define SCALE_ 0.25f        // 1/sqrt(16)
#define NEGINF (-1.0e9f)

typedef __attribute__((ext_vector_type(16))) _Float16 v16h;
typedef __attribute__((ext_vector_type(8)))  float    v8f;
typedef __attribute__((ext_vector_type(4)))  unsigned int v4u;
typedef __attribute__((ext_vector_type(8)))  unsigned int v8u;

union Frag16 {
  v16h v;
  float4 f4[2];
  _Float16 h[16];
};

// ---------------- WMMA fragment loaders (CDNA5 ISA §7.12.2 layouts) ----------------

// A-matrix 16x32 f16, row-major source (ld halfs per row).
// lane L: m=L&15, sel=L>>4 ; h[0..7]=K[sel*8 .. +7], h[8..15]=K[16+sel*8 .. +7]
__device__ inline v16h load_a_rm(const _Float16* p, int ld, int lane) {
  int m = lane & 15, sel = lane >> 4;
  Frag16 u;
  const _Float16* row = p + (long)m * ld;
  u.f4[0] = *reinterpret_cast<const float4*>(row + sel * 8);
  u.f4[1] = *reinterpret_cast<const float4*>(row + 16 + sel * 8);
  return u.v;
}

// A with only K=0..15 valid (d_k=16 padded to 32 with zeros)
__device__ inline v16h load_a_rm_pad16(const _Float16* p, int ld, int lane) {
  int m = lane & 15, sel = lane >> 4;
  Frag16 u;
  u.f4[0] = *reinterpret_cast<const float4*>(p + (long)m * ld + sel * 8);
  u.f4[1] = make_float4(0.f, 0.f, 0.f, 0.f);
  return u.v;
}

// B-matrix 32x16 f16 from (N,K) row-major source (weights stored (out,in)).
// lane L: n=L&15, sel=L>>4 ; h[0..7] = src[n*ld + sel*8 + 0..7], h[8..15] = src[n*ld + 16 + sel*8 + 0..7]
__device__ inline v16h load_b_nk(const _Float16* p, int ld, int lane) {
  int n = lane & 15, sel = lane >> 4;
  Frag16 u;
  const _Float16* row = p + (long)n * ld + sel * 8;
  u.f4[0] = *reinterpret_cast<const float4*>(row);
  u.f4[1] = *reinterpret_cast<const float4*>(row + 16);
  return u.v;
}

// B with only K=0..15 valid (keys' d_k=16 padded to 32)
__device__ inline v16h load_b_nk_pad16(const _Float16* p, int ld, int lane) {
  int n = lane & 15, sel = lane >> 4;
  Frag16 u;
  u.f4[0] = *reinterpret_cast<const float4*>(p + (long)n * ld + sel * 8);
  u.f4[1] = make_float4(0.f, 0.f, 0.f, 0.f);
  return u.v;
}

// B-matrix 32x16 f16 from (K,N) row-major source (e.g. V: rows=keys, cols=d_k)
__device__ inline v16h load_b_km(const _Float16* p, int ld, int lane) {
  int n = lane & 15, sel = lane >> 4;
  Frag16 u;
#pragma unroll
  for (int v = 0; v < 8; ++v) {
    int k = (v < 4 ? 0 : 8) + sel * 8 + 2 * v;
    u.h[2 * v]     = p[(long)k * ld + n];
    u.h[2 * v + 1] = p[(long)(k + 1) * ld + n];
  }
  return u.v;
}

__device__ inline v8f wmma_f16(v16h a, v16h b, v8f c) {
  return __builtin_amdgcn_wmma_f32_16x16x32_f16(false, a, false, b, (short)0, c, false, false);
}

__device__ inline float half_reduce_max(float v) {
  v = fmaxf(v, __shfl_xor(v, 1));
  v = fmaxf(v, __shfl_xor(v, 2));
  v = fmaxf(v, __shfl_xor(v, 4));
  v = fmaxf(v, __shfl_xor(v, 8));
  return v;
}
__device__ inline float half_reduce_sum(float v) {
  v += __shfl_xor(v, 1);
  v += __shfl_xor(v, 2);
  v += __shfl_xor(v, 4);
  v += __shfl_xor(v, 8);
  return v;
}

// ---------------- small utility kernels ----------------

__global__ void k_f2h(const float* __restrict__ s, _Float16* __restrict__ d, long n) {
  long i = (long)blockIdx.x * blockDim.x + threadIdx.x;
  if (i < n) d[i] = (_Float16)s[i];
}

__global__ void k_zero8(unsigned char* __restrict__ p, long n) {
  long i = (long)blockIdx.x * blockDim.x + threadIdx.x;
  if (i < n) p[i] = 0;
}

// h = coords @ in_W^T + in_b ; row 1024 = gnode ; rows 1025.. = 0
__global__ void k_embed(const float* __restrict__ coords, const float* __restrict__ inW,
                        const float* __restrict__ inb, const float* __restrict__ gnode,
                        float* __restrict__ hg) {
  long idx = (long)blockIdx.x * blockDim.x + threadIdx.x;
  if (idx >= (long)Bsz * MP * Dm) return;
  int d = (int)(idx % Dm);
  long t = idx / Dm;
  int i = (int)(t % MP);
  int b = (int)(t / MP);
  float v;
  if (i < Nn) {
    const float* c = coords + ((long)b * Nn + i) * 2;
    v = inW[d * 2 + 0] * c[0] + inW[d * 2 + 1] * c[1] + inb[d];
  } else if (i == Nn) {
    v = gnode[d];
  } else {
    v = 0.f;
  }
  hg[idx] = v;
}

// bucket[b,i,j] = clip(int(dist*32),0,31), coords_ext row 1024 = (0,0)
__global__ void k_bucket(const float* __restrict__ coords, unsigned char* __restrict__ bucket) {
  long idx = (long)blockIdx.x * blockDim.x + threadIdx.x;
  if (idx >= (long)Bsz * N1 * N1) return;
  int j = (int)(idx % N1);
  long t = idx / N1;
  int i = (int)(t % N1);
  int b = (int)(t / N1);
  float xi = 0.f, yi = 0.f, xj = 0.f, yj = 0.f;
  if (i < Nn) { xi = coords[((long)b * Nn + i) * 2]; yi = coords[((long)b * Nn + i) * 2 + 1]; }
  if (j < Nn) { xj = coords[((long)b * Nn + j) * 2]; yj = coords[((long)b * Nn + j) * 2 + 1]; }
  float dx = xi - xj, dy = yi - yj;
  float dist = sqrtf(dx * dx + dy * dy);
  int bt = (int)(dist * (float)NB);
  bt = bt < 0 ? 0 : (bt > NB - 1 ? NB - 1 : bt);
  bucket[idx] = (unsigned char)bt;
}

// g (row 1024) += mean over rows 0..1023
__global__ void k_gupdate(float* __restrict__ hg) {
  int b = blockIdx.x, d = threadIdx.x;
  long base = (long)b * MP * Dm;
  float s = 0.f;
  for (int r = 0; r < Nn; ++r) s += hg[base + (long)r * Dm + d];
  hg[base + (long)Nn * Dm + d] += s * (1.f / (float)Nn);
}

__global__ void k_pool(const float* __restrict__ hg, float* __restrict__ pool) {
  int b = blockIdx.x, d = threadIdx.x;
  long base = (long)b * MP * Dm;
  float s = 0.f;
  for (int r = 0; r < Nn; ++r) s += hg[base + (long)r * Dm + d];
  pool[b * Dm + d] = s * (1.f / (float)Nn);
}

__global__ void k_sq(const float* __restrict__ hg, float* __restrict__ sq) {
  int idx = blockIdx.x * blockDim.x + threadIdx.x;
  if (idx >= Bsz * Nn) return;
  int i = idx % Nn, b = idx / Nn;
  const float* row = hg + (long)b * MP * Dm + (long)i * Dm;
  float s = 0.f;
  for (int d = 0; d < Dm; ++d) s += row[d] * row[d];
  sq[idx] = s;
}

// kNN (k=10) over pairwise squared distances of h ; symmetric adjacency bytes.
// Query row h_i is staged into LDS with ASYNC global->LDS loads (ASYNCcnt path).
__global__ void k_knn(const float* __restrict__ hg, const float* __restrict__ sq,
                      unsigned char* __restrict__ adj) {
  __shared__ __align__(16) float hi[Dm];
  __shared__ float d2[Nn];
  __shared__ float rv[256];
  __shared__ int   ri[256];
  int i = blockIdx.x, b = blockIdx.y, tid = threadIdx.x;
  const float* hb = hg + (long)b * MP * Dm;
  if (tid < 32) {
    // 128 floats = 512 B : one b128 async transfer per lane of wave 0
    unsigned ldsa = (unsigned)(size_t)(&hi[0]) + (unsigned)(tid * 16);
    const char* gsrc = (const char*)(hb + (long)i * Dm) + tid * 16;
    asm volatile("global_load_async_to_lds_b128 %0, %1, off"
                 :: "v"(ldsa), "v"(gsrc) : "memory");
    asm volatile("s_wait_asynccnt 0x0" ::: "memory");
  }
  __syncthreads();
  float sqi = sq[b * Nn + i];
  for (int j = tid; j < Nn; j += 256) {
    const float* hj = hb + (long)j * Dm;
    float dot = 0.f;
    for (int d = 0; d < Dm; ++d) dot += hi[d] * hj[d];
    float v = sqi + sq[b * Nn + j] - 2.f * dot;
    d2[j] = (j == i) ? 3.4e38f : v;
  }
  __syncthreads();
  for (int it = 0; it < 10; ++it) {
    float bv = 3.4e38f; int bj = 0;
    for (int j = tid; j < Nn; j += 256) {
      float v = d2[j];
      if (v < bv) { bv = v; bj = j; }
    }
    rv[tid] = bv; ri[tid] = bj;
    __syncthreads();
    for (int s = 128; s > 0; s >>= 1) {
      if (tid < s && rv[tid + s] < rv[tid]) { rv[tid] = rv[tid + s]; ri[tid] = ri[tid + s]; }
      __syncthreads();
    }
    if (tid == 0) {
      int jm = ri[0];
      adj[(long)b * Nn * Nn + (long)i * Nn + jm] = 1;
      adj[(long)b * Nn * Nn + (long)jm * Nn + i] = 1;
      d2[jm] = 3.4e38f;
    }
    __syncthreads();
  }
}

// qe[b,h,i,t] = sum_d Q[b,i,h*16+d] * emb[t, h*16+d]
__global__ void k_qe(const _Float16* __restrict__ q16, const float* __restrict__ emb,
                     float* __restrict__ qe) {
  long idx = (long)blockIdx.x * blockDim.x + threadIdx.x;
  if (idx >= (long)Bsz * NH * MP * NB) return;
  int t = (int)(idx % NB);
  long r = idx / NB;
  int i = (int)(r % MP);
  long r2 = r / MP;
  int h = (int)(r2 % NH);
  int b = (int)(r2 / NH);
  const _Float16* q = q16 + (long)b * MP * Dm + (long)i * Dm + h * DK;
  const float* e = emb + (long)t * Dm + h * DK;
  float s = 0.f;
#pragma unroll
  for (int d = 0; d < DK; ++d) s += (float)q[d] * e[d];
  qe[idx] = s;
}

// ---------------- WMMA GEMM with TDM-staged A strip ----------------
// C = A(MxK,f16,row-major) @ Bw(NxK,f16,row-major)^T (+bias)(+relu)
// Block = 8 waves. The 16-row A strip (contiguous, lda==K) is DMA'd into LDS
// once per block by the Tensor Data Mover; each wave then owns one 16-wide
// N-tile, reading A from LDS and streaming its weight strip from L2.
__global__ void k_gemm(const _Float16* __restrict__ A, const _Float16* __restrict__ Bw,
                       const float* __restrict__ bias, float* __restrict__ C32,
                       _Float16* __restrict__ C16,
                       int K, int Nc, long long aStride, long long cStride, int relu) {
  __shared__ __align__(16) _Float16 Ash[16 * 512];   // up to K=512
  int tid = threadIdx.x;
  int lane = tid & 31;
  int wv = tid >> 5;                                  // 0..7
  int mt = blockIdx.y, b = blockIdx.z;
  int nt = blockIdx.x * 8 + wv;
  const _Float16* Ab = A + (long long)b * aStride + (long long)mt * 16 * K;
  const _Float16* Bb = Bw + (long long)nt * 16 * K;

  if (wv == 0) {
    // Tensor DMA descriptor (D# groups 0/1, CDNA5 ISA):
    // 1-D tile of 16*K halfs, data_size = 8 bytes.
    unsigned n8   = (unsigned)((16 * K) >> 2);        // # of 8-byte elements
    unsigned long long ga = (unsigned long long)(size_t)Ab;
    unsigned ldsa = (unsigned)(size_t)(&Ash[0]);
    v4u g0;
    g0.x = 1u;                                        // count=1 (valid descriptor)
    g0.y = ldsa;                                      // lds_addr
    g0.z = (unsigned)(ga & 0xffffffffu);              // global_addr[31:0]
    g0.w = ((unsigned)((ga >> 32) & 0x01ffffffu)) | (2u << 30);  // addr[56:32] | type=2
    v8u g1;
    g1.s0 = 3u << 16;                                 // wg_mask=0, data_size=3 (8B)
    g1.s1 = n8 << 16;                                 // tensor_dim0[15:0] (<<16)
    g1.s2 = (1u << 16) | (n8 >> 16);                  // tensor_dim0[31:16] | tensor_dim1 lo
    g1.s3 = n8 << 16;                                 // tile_dim0
    g1.s4 = 1u;                                       // tile_dim1 = 1
    g1.s5 = n8;                                       // tensor_dim0_stride[31:0]
    g1.s6 = 0u;
    g1.s7 = 0u;
    asm volatile("tensor_load_to_lds %0, %1" :: "s"(g0), "s"(g1) : "memory");
    __builtin_amdgcn_s_wait_tensorcnt(0);
  }
  __syncthreads();

  v8f acc = {0.f, 0.f, 0.f, 0.f, 0.f, 0.f, 0.f, 0.f};
  int ksteps = K >> 5;
  const _Float16* Asp = &Ash[0];
  for (int kt = 0; kt < ksteps; ++kt) {
    if (kt + 1 < ksteps) __builtin_prefetch(Bb + (kt + 1) * 32, 0, 3);
    v16h a = load_a_rm(Asp + kt * 32, K, lane);
    v16h bf = load_b_nk(Bb + kt * 32, K, lane);
    acc = wmma_f16(a, bf, acc);
  }
  int n = lane & 15, mb = (lane >> 4) * 8;
  float bv = bias ? bias[nt * 16 + n] : 0.f;
#pragma unroll
  for (int r = 0; r < 8; ++r) {
    float v = acc[r] + bv;
    if (relu) v = fmaxf(v, 0.f);
    long long ci = (long long)b * cStride + (long long)(mt * 16 + mb + r) * Nc + nt * 16 + n;
    if (C32) C32[ci] = v;
    if (C16) C16[ci] = (_Float16)v;
  }
}

// ---------------- fused attention (online softmax, per 16-query tile per head) ----------------
__device__ inline float extra_term(int b, int h, int i, int j,
                                   const float* __restrict__ qe,
                                   const unsigned char* __restrict__ bucket,
                                   const unsigned char* __restrict__ adj) {
  if (i > Nn || j > Nn) return NEGINF;   // padded rows/cols
  unsigned char t = bucket[(long)b * N1 * N1 + (long)i * N1 + j];
  float rel = qe[(((long)b * NH + h) * MP + i) * NB + t];
  float mval;
  if (i == Nn || j == Nn) {
    mval = 0.f;                           // global node fully connected
  } else {
    bool conn = (i == j) || (i == 0) || (j == 0) ||
                (adj[(long)b * Nn * Nn + (long)i * Nn + j] != 0);
    if (!conn) {
      int di = i - j; if (di < 0) di = -di;
      conn = ((i >> 7) == (j >> 7)) && (di <= 11);   // hierarchical band, blocks of 128
    }
    mval = conn ? 0.f : NEGINF;
  }
  return rel + mval;
}

__global__ void k_attn(const _Float16* __restrict__ q16, const _Float16* __restrict__ k16,
                       const _Float16* __restrict__ v16, const float* __restrict__ qe,
                       const unsigned char* __restrict__ bucket,
                       const unsigned char* __restrict__ adj,
                       _Float16* __restrict__ ctx16) {
  __shared__ __align__(16) _Float16 Psh[16 * 32];
  int lane = threadIdx.x;
  int qt = blockIdx.x;
  int bh = blockIdx.y;
  int b = bh >> 3, h = bh & 7;
  const long bb = (long)b * MP * Dm;

  v16h aq = load_a_rm_pad16(q16 + bb + (long)qt * 16 * Dm + h * DK, Dm, lane);

  int n = lane & 15, mb = (lane >> 4) * 8;
  float mrow[8], lrow[8];
  v8f acc = {0.f, 0.f, 0.f, 0.f, 0.f, 0.f, 0.f, 0.f};
#pragma unroll
  for (int r = 0; r < 8; ++r) { mrow[r] = -1e30f; lrow[r] = 0.f; }

  for (int jp = 0; jp < MT / 2; ++jp) {   // 33 pairs of 16-key tiles
    int j0 = jp * 32;
    v16h kb0 = load_b_nk_pad16(k16 + bb + (long)j0 * Dm + h * DK, Dm, lane);
    v16h kb1 = load_b_nk_pad16(k16 + bb + (long)(j0 + 16) * Dm + h * DK, Dm, lane);
    v8f z = {0.f, 0.f, 0.f, 0.f, 0.f, 0.f, 0.f, 0.f};
    v8f s0 = wmma_f16(aq, kb0, z);
    v8f s1 = wmma_f16(aq, kb1, z);

#pragma unroll
    for (int r = 0; r < 8; ++r) {
      int i = qt * 16 + mb + r;
      float x0 = s0[r] * SCALE_ + extra_term(b, h, i, j0 + n, qe, bucket, adj);
      float x1 = s1[r] * SCALE_ + extra_term(b, h, i, j0 + 16 + n, qe, bucket, adj);
      float mx = half_reduce_max(fmaxf(x0, x1));
      float nm = fmaxf(mrow[r], mx);
      float corr = __expf(mrow[r] - nm);
      float p0 = __expf(x0 - nm);
      float p1 = __expf(x1 - nm);
      float rs = half_reduce_sum(p0 + p1);
      lrow[r] = lrow[r] * corr + rs;
      mrow[r] = nm;
      acc[r] *= corr;
      Psh[(mb + r) * 32 + n] = (_Float16)p0;
      Psh[(mb + r) * 32 + 16 + n] = (_Float16)p1;
    }
    __syncthreads();
    v16h pa = load_a_rm(Psh, 32, lane);
    v16h vb = load_b_km(v16 + bb + (long)j0 * Dm + h * DK, Dm, lane);
    acc = wmma_f16(pa, vb, acc);
    __syncthreads();
  }

#pragma unroll
  for (int r = 0; r < 8; ++r) {
    int row = qt * 16 + mb + r;
    float o = (lrow[r] > 0.f) ? acc[r] / lrow[r] : 0.f;
    ctx16[bb + (long)row * Dm + h * DK + n] = (_Float16)o;
  }
}

// ---------------- instance norm over node axis (1025 rows), + residual, zero-pad rows ----------
__global__ void k_inorm(const float* __restrict__ x, const float* __restrict__ resid,
                        const float* __restrict__ w, const float* __restrict__ bb,
                        float* __restrict__ out32, _Float16* __restrict__ out16) {
  int b = blockIdx.x, d = threadIdx.x;
  long base = (long)b * MP * Dm;
  float s = 0.f, ss = 0.f;
  for (int r = 0; r < N1; ++r) {
    float v = x[base + (long)r * Dm + d] + resid[base + (long)r * Dm + d];
    s += v; ss += v * v;
  }
  const float invn = 1.f / (float)N1;
  float mu = s * invn;
  float var = ss * invn - mu * mu;
  float inv = rsqrtf(var + 1e-5f);
  float wd = w[d], bd = bb[d];
  for (int r = 0; r < N1; ++r) {
    long o = base + (long)r * Dm + d;
    float v = x[o] + resid[o];
    float y = (v - mu) * inv * wd + bd;
    out32[o] = y;
    if (out16) out16[o] = (_Float16)y;
  }
  for (int r = N1; r < MP; ++r) {
    long o = base + (long)r * Dm + d;
    out32[o] = 0.f;
    if (out16) out16[o] = (_Float16)0.f;
  }
}

__global__ void k_pmean(const float* __restrict__ pools, float* __restrict__ pm) {
  int idx = threadIdx.x;            // B*D = 256
  if (idx >= Bsz * Dm) return;
  pm[idx] = (pools[idx] + pools[Bsz * Dm + idx] + pools[2 * Bsz * Dm + idx]) * (1.f / 3.f);
}

__global__ void k_addpm(const float* __restrict__ hg, const float* __restrict__ pm,
                        _Float16* __restrict__ tmp16) {
  long idx = (long)blockIdx.x * blockDim.x + threadIdx.x;
  if (idx >= (long)Bsz * Nn * Dm) return;
  int d = (int)(idx % Dm);
  long t = idx / Dm;
  int i = (int)(t % Nn);
  int b = (int)(t / Nn);
  tmp16[idx] = (_Float16)(hg[(long)b * MP * Dm + (long)i * Dm + d] + pm[b * Dm + d]);
}

// ---------------- host orchestration ----------------
static inline int nblk(long n, int t) { return (int)((n + t - 1) / t); }

extern "C" void kernel_launch(void* const* d_in, const int* in_sizes, int n_in,
                              void* d_out, int out_size, void* d_ws, size_t ws_size,
                              hipStream_t stream) {
  const float* coords = (const float*)d_in[0];
  const float* in_W   = (const float*)d_in[1];
  const float* in_b   = (const float*)d_in[2];
  const float* gnode  = (const float*)d_in[3];
  const float* Wq     = (const float*)d_in[4];
  const float* Wk     = (const float*)d_in[5];
  const float* Wv     = (const float*)d_in[6];
  const float* Wo     = (const float*)d_in[7];
  const float* emb    = (const float*)d_in[8];
  const float* W1     = (const float*)d_in[9];
  const float* b1     = (const float*)d_in[10];
  const float* W2     = (const float*)d_in[11];
  const float* b2     = (const float*)d_in[12];
  const float* n1_w   = (const float*)d_in[13];
  const float* n1_b   = (const float*)d_in[14];
  const float* n2_w   = (const float*)d_in[15];
  const float* n2_b   = (const float*)d_in[16];
  const float* out_W  = (const float*)d_in[17];
  const float* out_b  = (const float*)d_in[18];
  float* out = (float*)d_out;

  // bump allocator over workspace
  char* ws = (char*)d_ws;
  size_t off = 0;
  auto alloc = [&](size_t bytes) -> void* {
    void* p = ws + off;
    off = (off + bytes + 255) & ~(size_t)255;
    return p;
  };

  float*     hgA    = (float*)alloc((size_t)Bsz * MP * Dm * 4);
  float*     hgB    = (float*)alloc((size_t)Bsz * MP * Dm * 4);
  _Float16*  hg16   = (_Float16*)alloc((size_t)Bsz * MP * Dm * 2);
  _Float16*  q16    = (_Float16*)alloc((size_t)Bsz * MP * Dm * 2);
  _Float16*  k16    = (_Float16*)alloc((size_t)Bsz * MP * Dm * 2);
  _Float16*  v16    = (_Float16*)alloc((size_t)Bsz * MP * Dm * 2);
  _Float16*  ctx16  = (_Float16*)alloc((size_t)Bsz * MP * Dm * 2);
  _Float16*  h116   = (_Float16*)alloc((size_t)Bsz * MP * Dm * 2);
  _Float16*  ffn116 = (_Float16*)alloc((size_t)Bsz * MP * FF * 2);
  _Float16*  tmp16  = (_Float16*)alloc((size_t)Bsz * Nn * Dm * 2);
  float*     wo32   = (float*)alloc((size_t)Bsz * MP * Dm * 4);
  float*     h132   = (float*)alloc((size_t)Bsz * MP * Dm * 4);
  float*     ffn232 = (float*)alloc((size_t)Bsz * MP * Dm * 4);
  float*     qe32   = (float*)alloc((size_t)Bsz * NH * MP * NB * 4);
  float*     sqb    = (float*)alloc((size_t)Bsz * Nn * 4);
  float*     pools  = (float*)alloc((size_t)NL * Bsz * Dm * 4);
  unsigned char* adj     = (unsigned char*)alloc((size_t)Bsz * Nn * Nn);
  unsigned char* bucketb = (unsigned char*)alloc((size_t)Bsz * N1 * N1);
  _Float16*  wq16  = (_Float16*)alloc((size_t)NL * Dm * Dm * 2);
  _Float16*  wk16  = (_Float16*)alloc((size_t)NL * Dm * Dm * 2);
  _Float16*  wv16  = (_Float16*)alloc((size_t)NL * Dm * Dm * 2);
  _Float16*  wo16  = (_Float16*)alloc((size_t)NL * Dm * Dm * 2);
  _Float16*  w116  = (_Float16*)alloc((size_t)NL * FF * Dm * 2);
  _Float16*  w216  = (_Float16*)alloc((size_t)NL * Dm * FF * 2);
  _Float16*  outw16 = (_Float16*)alloc((size_t)Dm * Dm * 2);

  // weights fp32 -> fp16
  {
    long nw = (long)NL * Dm * Dm;
    k_f2h<<<nblk(nw, 256), 256, 0, stream>>>(Wq, wq16, nw);
    k_f2h<<<nblk(nw, 256), 256, 0, stream>>>(Wk, wk16, nw);
    k_f2h<<<nblk(nw, 256), 256, 0, stream>>>(Wv, wv16, nw);
    k_f2h<<<nblk(nw, 256), 256, 0, stream>>>(Wo, wo16, nw);
    long n1n = (long)NL * FF * Dm;
    k_f2h<<<nblk(n1n, 256), 256, 0, stream>>>(W1, w116, n1n);
    k_f2h<<<nblk(n1n, 256), 256, 0, stream>>>(W2, w216, n1n);
    long no = (long)Dm * Dm;
    k_f2h<<<nblk(no, 256), 256, 0, stream>>>(out_W, outw16, no);
  }

  // embed + buckets
  k_embed<<<nblk((long)Bsz * MP * Dm, 256), 256, 0, stream>>>(coords, in_W, in_b, gnode, hgA);
  k_bucket<<<nblk((long)Bsz * N1 * N1, 256), 256, 0, stream>>>(coords, bucketb);

  float* cur = hgA;
  float* nxt = hgB;
  const long long sHG = (long long)MP * Dm;
  const long long sFF = (long long)MP * FF;

  for (int l = 0; l < NL; ++l) {
    k_gupdate<<<Bsz, Dm, 0, stream>>>(cur);
    k_f2h<<<nblk((long)Bsz * MP * Dm, 256), 256, 0, stream>>>(cur, hg16, (long)Bsz * MP * Dm);

    k_zero8<<<nblk((long)Bsz * Nn * Nn, 256), 256, 0, stream>>>(adj, (long)Bsz * Nn * Nn);
    k_sq<<<nblk(Bsz * Nn, 256), 256, 0, stream>>>(cur, sqb);
    k_knn<<<dim3(Nn, Bsz), 256, 0, stream>>>(cur, sqb, adj);

    dim3 gp(Dm / 128, MT, Bsz);      // 1 x 66 x 2, 8 waves/block
    k_gemm<<<gp, 256, 0, stream>>>(hg16, wq16 + (long)l * Dm * Dm, nullptr, nullptr, q16,
                                   Dm, Dm, sHG, sHG, 0);
    k_gemm<<<gp, 256, 0, stream>>>(hg16, wk16 + (long)l * Dm * Dm, nullptr, nullptr, k16,
                                   Dm, Dm, sHG, sHG, 0);
    k_gemm<<<gp, 256, 0, stream>>>(hg16, wv16 + (long)l * Dm * Dm, nullptr, nullptr, v16,
                                   Dm, Dm, sHG, sHG, 0);

    k_qe<<<nblk((long)Bsz * NH * MP * NB, 256), 256, 0, stream>>>(q16, emb + (long)l * NB * Dm, qe32);

    k_attn<<<dim3(MT, Bsz * NH), 32, 0, stream>>>(q16, k16, v16, qe32, bucketb, adj, ctx16);

    k_gemm<<<gp, 256, 0, stream>>>(ctx16, wo16 + (long)l * Dm * Dm, nullptr, wo32, nullptr,
                                   Dm, Dm, sHG, sHG, 0);
    k_inorm<<<Bsz, Dm, 0, stream>>>(wo32, cur, n1_w + l * Dm, n1_b + l * Dm, h132, h116);

    dim3 g1(FF / 128, MT, Bsz);      // 4 x 66 x 2
    k_gemm<<<g1, 256, 0, stream>>>(h116, w116 + (long)l * FF * Dm, b1 + l * FF, nullptr, ffn116,
                                   Dm, FF, sHG, sFF, 1);
    dim3 g2(Dm / 128, MT, Bsz);
    k_gemm<<<g2, 256, 0, stream>>>(ffn116, w216 + (long)l * Dm * FF, b2 + l * Dm, ffn232, nullptr,
                                   FF, Dm, sFF, sHG, 0);
    k_inorm<<<Bsz, Dm, 0, stream>>>(ffn232, h132, n2_w + l * Dm, n2_b + l * Dm, nxt, nullptr);

    k_pool<<<Bsz, Dm, 0, stream>>>(nxt, pools + (long)l * Bsz * Dm);

    float* t = cur; cur = nxt; nxt = t;
  }

  // p_multi -> tail of d_out ; E_static -> head of d_out
  float* pm_out = out + (size_t)Bsz * Nn * Dm;
  k_pmean<<<1, 256, 0, stream>>>(pools, pm_out);
  k_addpm<<<nblk((long)Bsz * Nn * Dm, 256), 256, 0, stream>>>(cur, pm_out, tmp16);
  dim3 gf(Dm / 128, Nn / 16, Bsz);
  k_gemm<<<gf, 256, 0, stream>>>(tmp16, outw16, out_b, out, nullptr,
                                 Dm, Dm, (long long)Nn * Dm, (long long)Nn * Dm, 0);
  (void)in_sizes; (void)n_in; (void)out_size; (void)ws_size;
}
